// Attention_74002286510368
// MI455X (gfx1250) — matrix operand order repaired and checked by reading, not compiled, not run on hardware
//
#include <hip/hip_runtime.h>

// ---------------------------------------------------------------------------
// Problem constants (from reference setup_inputs)
// ---------------------------------------------------------------------------
#define BB   8
#define LD   128
#define LE   512
#define NENC 512
#define NDEC 512
#define NATT 256
#define ME   (BB * LE)   // 4096 rows of pe (transposed leading dim)

typedef __attribute__((ext_vector_type(16))) __bf16 v16bf;
typedef __attribute__((ext_vector_type(8)))  float  v8f;

union BfFrag {
    uint4  u[2];   // 32 bytes = 16 bf16
    v16bf  v;
};

// fast tanh: gfx1250 has V_TANH_F32 (single trans instruction)
#if __has_builtin(__builtin_amdgcn_tanhf)
__device__ __forceinline__ float t_tanh(float x) { return __builtin_amdgcn_tanhf(x); }
#else
__device__ __forceinline__ float t_tanh(float x) { return tanhf(x); }
#endif

// ---------------------------------------------------------------------------
// Kernel 0: fp32 -> bf16 (round to nearest even), grid-stride
// ---------------------------------------------------------------------------
__device__ __forceinline__ unsigned short f2bf(float f) {
    unsigned u = __float_as_uint(f);
    u += 0x7FFFu + ((u >> 16) & 1u);
    return (unsigned short)(u >> 16);
}

__global__ void cvt_f32_to_bf16(const float* __restrict__ in,
                                unsigned short* __restrict__ out, int n) {
    int i = blockIdx.x * blockDim.x + threadIdx.x;
    int stride = gridDim.x * blockDim.x;
    for (; i < n; i += stride) out[i] = f2bf(in[i]);
}

// ---------------------------------------------------------------------------
// Kernel 1: C = A[M,K] * W[N,K]^T (+ bias[n]) via v_wmma_f32_16x16x32_bf16
//   A, W bf16 row-major (K contiguous). One 16x16 C tile per wave, 4 waves
//   per block striding M. K mult of 32, M mult of 64, N mult of 16.
//   transC==0 : C[m*N + n]           (row-major,    used for pd)
//   transC==1 : C[n*M + m]           (transposed,   used for pe; each lane
//               stores 8 consecutive m -> two contiguous float4 stores)
//
// Fragment mapping (per CDNA5 ISA 7.12.2, 16-bit operands, wave32):
//   A 16x32:  lane L<16 -> row L,  halves 0..7 = K k..k+7,  8..15 = K k+16..k+23
//             lane L>=16 -> row L-16, halves 0..7 = K k+8..k+15, 8..15 = k+24..k+31
//   B 32x16:  lane L<16 -> col L,  halves 0..7 = K k..k+7,  8..15 = K k+16..k+23
//             lane L>=16 -> col L-16, halves 0..7 = K k+8..k+15, 8..15 = k+24..k+31
//   C 16x16 f32: vgpr v, lane L<16 -> C[v][L] ; lane L>=16 -> C[v+8][L-16]
// ---------------------------------------------------------------------------
__global__ __launch_bounds__(128)
void gemm_bf16_wmma(const unsigned short* __restrict__ A,
                    const unsigned short* __restrict__ W,
                    const float* __restrict__ bias,
                    float* __restrict__ C,
                    int M, int N, int K, int transC) {
    const int wave = threadIdx.x >> 5;
    const int lane = threadIdx.x & 31;
    const int mBase = (blockIdx.x * 4 + wave) * 16;
    const int nBase = blockIdx.y * 16;
    if (mBase >= M) return;

    const int row = lane & 15;     // A row within tile / B column within tile
    const int hi  = lane >> 4;     // lane-group select

    const unsigned short* aRow = A + (size_t)(mBase + row) * K + hi * 8;
    const unsigned short* wRow = W + (size_t)(nBase + row) * K + hi * 8;

    v8f acc = {};
    for (int k = 0; k < K; k += 32) {
        BfFrag a, b;
        a.u[0] = *(const uint4*)(aRow + k);        // K = k + hi*8 + (0..7)
        a.u[1] = *(const uint4*)(aRow + k + 16);   // K = k + hi*8 + 16 + (0..7)
        b.u[0] = *(const uint4*)(wRow + k);        // K = k + hi*8 + (0..7)
        b.u[1] = *(const uint4*)(wRow + k + 16);   // K = k + hi*8 + 16 + (0..7)
        acc = __builtin_amdgcn_wmma_f32_16x16x32_bf16(
                  false, a.v, false, b.v, (short)0, acc, false, false);
    }

    const int n  = lane & 15;
    const float bv = bias ? bias[nBase + n] : 0.0f;
    if (transC) {
        // C[n_global * M + m], lane's 8 outputs are consecutive m
        float* cOut = C + (size_t)(nBase + n) * M + mBase + hi * 8;
        float4 o0 = make_float4(acc[0] + bv, acc[1] + bv, acc[2] + bv, acc[3] + bv);
        float4 o1 = make_float4(acc[4] + bv, acc[5] + bv, acc[6] + bv, acc[7] + bv);
        *(float4*)(cOut)     = o0;
        *(float4*)(cOut + 4) = o1;
    } else {
        float* cOut = C + (size_t)(mBase + hi * 8) * N + nBase + n;
#pragma unroll
        for (int v = 0; v < 8; ++v)
            cOut[(size_t)v * N] = acc[v] + bv;
    }
}

// ---------------------------------------------------------------------------
// Kernel 2: fused scores + tanh + masked softmax + renormalize.
//   One block (256 threads) per (b,d). Thread t owns e = t and e = t+256.
//   peT is [NATT][B*LE]: at fixed n, lanes read consecutive e -> fully
//   coalesced 128B wave loads. W_att[n] / pd[b,d,n] broadcast from LDS.
//   scores[e] = b_att + sum_n W_att[n]*tanh(pd[b,d,n] + pe[b,e,n])
//   out = ssm_e / (sum(ssm) + EPS * sum(ss))   [== softmax*mask renormalized]
// ---------------------------------------------------------------------------
__global__ __launch_bounds__(256)
void scores_softmax(const float* __restrict__ peT,  // [NATT, B*LE]
                    const float* __restrict__ pd,   // [B*LD, NATT]
                    const float* __restrict__ mask, // [B, LE]
                    const float* __restrict__ Watt, // [NATT]
                    const float* __restrict__ batt, // [1]
                    float* __restrict__ out) {      // [B, LD, LE]
    const int bd = blockIdx.x;
    const int b  = bd / LD;
    const int d  = bd % LD;
    const int t  = threadIdx.x;

    __shared__ float sPd[NATT];
    __shared__ float sW[NATT];
    __shared__ float r1[256];
    __shared__ float r2[256];

    sPd[t] = pd[((size_t)b * LD + d) * NATT + t];
    sW[t]  = Watt[t];
    __syncthreads();

    const float ba = batt[0];
    const int ei0 = t, ei1 = t + 256;
    const float* p0 = peT + (size_t)b * LE + ei0;  // column e=ei0, step ME per n
    const float* p1 = peT + (size_t)b * LE + ei1;

    float s0 = ba, s1 = ba;
    for (int n = 0; n < NATT; n += 4) {
        float4 w  = *(const float4*)(sW + n);
        float4 dd = *(const float4*)(sPd + n);
        // coalesced single-dword wave loads, one line pair per n
        float x00 = p0[0 * ME], x01 = p0[1 * ME], x02 = p0[2 * ME], x03 = p0[3 * ME];
        float x10 = p1[0 * ME], x11 = p1[1 * ME], x12 = p1[2 * ME], x13 = p1[3 * ME];
        // hint L2->WGP prefetch of lines ~2 iterations ahead (global_prefetch_b8)
        __builtin_prefetch(p0 + 8 * ME, 0, 1);
        __builtin_prefetch(p1 + 8 * ME, 0, 1);
        p0 += 4 * ME;
        p1 += 4 * ME;
        s0 = fmaf(w.x, t_tanh(dd.x + x00), s0);
        s0 = fmaf(w.y, t_tanh(dd.y + x01), s0);
        s0 = fmaf(w.z, t_tanh(dd.z + x02), s0);
        s0 = fmaf(w.w, t_tanh(dd.w + x03), s0);
        s1 = fmaf(w.x, t_tanh(dd.x + x10), s1);
        s1 = fmaf(w.y, t_tanh(dd.y + x11), s1);
        s1 = fmaf(w.z, t_tanh(dd.z + x12), s1);
        s1 = fmaf(w.w, t_tanh(dd.w + x13), s1);
    }

    // --- max reduction over 512 scores ---
    r1[t] = fmaxf(s0, s1);
    __syncthreads();
#pragma unroll
    for (int s = 128; s > 0; s >>= 1) {
        if (t < s) r1[t] = fmaxf(r1[t], r1[t + s]);
        __syncthreads();
    }
    const float mx = r1[0];
    __syncthreads();

    // --- exp, mask, paired sum reductions ---
    const float ex0 = __expf(s0 - mx);
    const float ex1 = __expf(s1 - mx);
    const float m0  = mask[(size_t)b * LE + ei0];
    const float m1  = mask[(size_t)b * LE + ei1];
    const float sm0 = ex0 * m0;
    const float sm1 = ex1 * m1;
    r1[t] = ex0 + ex1;
    r2[t] = sm0 + sm1;
    __syncthreads();
#pragma unroll
    for (int s = 128; s > 0; s >>= 1) {
        if (t < s) { r1[t] += r1[t + s]; r2[t] += r2[t + s]; }
        __syncthreads();
    }
    // out = (ssm/S) / (sum(ssm)/S + EPS) == ssm / (sum(ssm) + EPS*S)
    const float inv = 1.0f / (r2[0] + 1e-8f * r1[0]);

    float* o = out + ((size_t)b * LD + d) * LE;
    o[ei0] = sm0 * inv;
    o[ei1] = sm1 * inv;
}

// ---------------------------------------------------------------------------
// Launch
// ---------------------------------------------------------------------------
extern "C" void kernel_launch(void* const* d_in, const int* in_sizes, int n_in,
                              void* d_out, int out_size, void* d_ws, size_t ws_size,
                              hipStream_t stream) {
    (void)in_sizes; (void)n_in; (void)out_size; (void)ws_size;

    const float* h_e   = (const float*)d_in[0];  // [B, LE, NENC]
    const float* h_d   = (const float*)d_in[1];  // [B, LD, NDEC]
    const float* mask  = (const float*)d_in[2];  // [B, LE]
    const float* W_en  = (const float*)d_in[3];  // [NATT, NENC]
    const float* b_en  = (const float*)d_in[4];  // [NATT]
    const float* W_de  = (const float*)d_in[5];  // [NATT, NDEC]
    const float* W_att = (const float*)d_in[6];  // [1, NATT]
    const float* b_att = (const float*)d_in[7];  // [1]
    float* out = (float*)d_out;

    // workspace layout (all offsets 256B-aligned)
    char* ws = (char*)d_ws;
    const size_t nHe = (size_t)BB * LE * NENC;     // 2,097,152
    const size_t nHd = (size_t)BB * LD * NDEC;     //   524,288
    const size_t nWe = (size_t)NATT * NENC;        //   131,072
    const size_t nWd = (size_t)NATT * NDEC;        //   131,072
    unsigned short* heB = (unsigned short*)(ws);
    unsigned short* hdB = (unsigned short*)(ws + 4194304);
    unsigned short* weB = (unsigned short*)(ws + 5242880);
    unsigned short* wdB = (unsigned short*)(ws + 5505024);
    float*          peT = (float*)(ws + 5767168); // [NATT, B*LE] fp32 (transposed)
    float*          pd  = (float*)(ws + 9961472); // [B*LD, NATT] fp32

    // 0) down-convert fp32 -> bf16
    cvt_f32_to_bf16<<<dim3(2048), dim3(256), 0, stream>>>(h_e,  heB, (int)nHe);
    cvt_f32_to_bf16<<<dim3(1024), dim3(256), 0, stream>>>(h_d,  hdB, (int)nHd);
    cvt_f32_to_bf16<<<dim3(256),  dim3(256), 0, stream>>>(W_en, weB, (int)nWe);
    cvt_f32_to_bf16<<<dim3(256),  dim3(256), 0, stream>>>(W_de, wdB, (int)nWd);

    // 1) projections via WMMA
    //    peT = (h_e @ W_en^T + b_en)^T : M = B*LE = 4096, N = NATT, K = NENC
    gemm_bf16_wmma<<<dim3(ME / 64, NATT / 16), dim3(128), 0, stream>>>(
        heB, weB, b_en, peT, ME, NATT, NENC, /*transC=*/1);
    //    pd  = h_d @ W_de^T            : M = B*LD = 1024, N = NATT, K = NDEC
    gemm_bf16_wmma<<<dim3((BB * LD) / 64, NATT / 16), dim3(128), 0, stream>>>(
        hdB, wdB, nullptr, pd, BB * LD, NATT, NDEC, /*transC=*/0);

    // 2) fused tanh-score + masked softmax + renormalize (trans-pipe bound)
    scores_softmax<<<dim3(BB * LD), dim3(256), 0, stream>>>(
        peT, pd, mask, W_att, b_att, out);
}
